// DeconvCapsuleLayer_29154238005458
// MI455X (gfx1250) — compile-verified, matrix operand order checked by reading, not run on hardware
//
#include <hip/hip_runtime.h>

typedef __attribute__((ext_vector_type(16))) _Float16 v16h;
typedef __attribute__((ext_vector_type(8)))  _Float16 v8h;
typedef __attribute__((ext_vector_type(8)))  float    v8f;
typedef __attribute__((ext_vector_type(4)))  float    v4f;
typedef __attribute__((ext_vector_type(2)))  float    v2f;
typedef __attribute__((ext_vector_type(4)))  unsigned int u32x4;
typedef __attribute__((ext_vector_type(8)))  unsigned int u32x8;

// Problem constants
#define NB   8
#define HIN  56
#define ICN  8
#define IAN  32
#define OCN  4
#define OAN  16
#define H2   112
#define NCH  64      // OC*OA
#define NPIX (H2*H2) // 12544

// ---------------------------------------------------------------------------
// Kernel 1: relayout input (B,H,W,IC,IA) f32 -> xh (B,IC,H,W,IA) f16
// ---------------------------------------------------------------------------
__global__ void k_pack_x(const float* __restrict__ in, _Float16* __restrict__ xh) {
    int idx = blockIdx.x * blockDim.x + threadIdx.x;   // exact: 6,422,528
    int a  = idx & 31;
    int t  = idx >> 5;
    int x  = t % HIN; t /= HIN;
    int y  = t % HIN; t /= HIN;
    int ic = t & 7;
    int b  = t >> 3;
    float v = in[(((b * HIN + y) * HIN + x) * ICN + ic) * IAN + a];
    xh[idx] = (_Float16)v;
}

// ---------------------------------------------------------------------------
// Kernel 2: swizzle W (kh,kw,64,32) f32 into per-lane WMMA B fragments (f16).
// Fragment element (khkw, nt, lane, e) holds W[khkw, n = nt*16 + lane%16,
// k = 8*(lane>=16) + 16*(e>=8) + e%8]  -- matching the 16-bit 16x32 K layout.
// ---------------------------------------------------------------------------
__global__ void k_pack_w(const float* __restrict__ W, _Float16* __restrict__ bf) {
    int tid  = blockIdx.x * blockDim.x + threadIdx.x;  // exact: 32768
    int e    = tid & 15;
    int lane = (tid >> 4) & 31;
    int nt   = (tid >> 9) & 3;
    int khkw = tid >> 11;                              // kh*4+kw
    int k = ((lane & 16) ? 8 : 0) + ((e & 8) ? 16 : 0) + (e & 7);
    int n = nt * 16 + (lane & 15);
    bf[tid] = (_Float16)W[(khkw * NCH + n) * IAN + k];
}

// ---------------------------------------------------------------------------
// Kernel 3: deconv as WMMA GEMM. One wave = 16 same-parity output pixels of
// one row: M=16, N=64, K=128 (2x2 taps x 32 ic). 16x v_wmma_f32_16x16x32_f16.
// The 64KB B-fragment table is DMA'd into LDS once per workgroup by the
// Tensor Data Mover (tensor_load_to_lds, TENSORcnt), then WMMA B operands
// come from LDS (ds_load_b128) instead of per-wave global reads.
// ---------------------------------------------------------------------------
__global__ void __launch_bounds__(256) k_votes(const _Float16* __restrict__ xh,
                                               const _Float16* __restrict__ bfrag,
                                               float* __restrict__ votes) {
    __shared__ _Float16 bsh[32768];                    // 64 KB B-fragment table
    __shared__ float    dstage[8][16 * NCH];           // 32 KB D staging per wave
    const int lane = threadIdx.x & 31;
    const int wid  = threadIdx.x >> 5;

    // ---- TDM: one 64KB 1-D tile load, issued by wave 0 only ----
    if (wid == 0) {
        unsigned long long ga = (unsigned long long)(uintptr_t)bfrag;
        unsigned ldsa = (unsigned)(uintptr_t)&bsh[0];  // LDS byte offset (flat addr[31:0])
        u32x4 g0;
        g0[0] = 1u;                                    // count=1, user descriptor
        g0[1] = ldsa;                                  // lds_addr
        g0[2] = (unsigned)ga;                          // global_addr[31:0]
        g0[3] = ((unsigned)(ga >> 32) & 0x01FFFFFFu)   // global_addr[56:32]
              | (2u << 30);                            // type = 2 ("image")
        u32x8 g1;
        g1[0] = 3u << 16;                              // data_size = 8 bytes
        g1[1] = 0x2000u << 16;                         // tensor_dim0 = 8192 elems (64KB)
        g1[2] = 1u << 16;                              // tensor_dim1 = 1
        g1[3] = 0x2000u << 16;                         // tile_dim0 = 8192
        g1[4] = 0;                                     // tile_dim1/2 unused
        g1[5] = 8192u;                                 // tensor_dim0_stride
        g1[6] = 0;
        g1[7] = 0;
        asm volatile("tensor_load_to_lds %0, %1" :: "s"(g0), "s"(g1) : "memory");
        __builtin_amdgcn_s_wait_tensorcnt(0);
    }
    __syncthreads();

    int flat = blockIdx.x * 8 + wid;                   // 0..57343
    int tile = flat & 7;
    int oy   = (flat >> 3) % H2;
    int bic  = flat / (8 * H2);
    int p    = tile >> 2;                              // ox parity
    int j0   = (tile & 3) * 16;                        // ox = p + 2*j

    // transposed-conv tap geometry: oy+1 = 2*iy + kh, ox+1 = 2*ix + kw
    int kh0 = (oy + 1) & 1;
    int iy0 = (oy + 1 - kh0) >> 1;
    int kw0 = (p + 1) & 1;
    int dx0 = (p + 1) >> 1;                            // ix = j + dx0 (kw0), -1 for kw0+2

    const int mrow = lane & 15;
    const int j    = j0 + mrow;
    const int chb  = (lane & 16) ? 8 : 0;              // K-chunk base per ISA A layout

    v8f acc0 = {}, acc1 = {}, acc2 = {}, acc3 = {};

    #pragma unroll
    for (int t = 0; t < 4; ++t) {
        int kh = kh0 + 2 * (t >> 1);
        int kw = kw0 + 2 * (t & 1);
        int iy = iy0 - (t >> 1);
        int ix = j + dx0 - (t & 1);
        bool valid = (iy >= 0) && (iy < HIN) && (ix >= 0) && (ix < HIN) && (j < HIN);

        v16h a = {};
        if (valid) {
            const _Float16* src = xh + (((bic * HIN + iy) * HIN + ix) * IAN + chb);
            v8h lo = *(const v8h*)(src);        // channels chb .. chb+7
            v8h hi = *(const v8h*)(src + 16);   // channels chb+16 .. chb+23
            #pragma unroll
            for (int e = 0; e < 8; ++e) { a[e] = lo[e]; a[e + 8] = hi[e]; }
        }
        int khkw = kh * 4 + kw;
        const _Float16* bp = &bsh[((khkw * 4) * 32 + lane) * 16];
        v16h b0 = *(const v16h*)(bp);           // nt stride = 512 halfs
        v16h b1 = *(const v16h*)(bp + 512);
        v16h b2 = *(const v16h*)(bp + 1024);
        v16h b3 = *(const v16h*)(bp + 1536);
        acc0 = __builtin_amdgcn_wmma_f32_16x16x32_f16(false, a, false, b0, (short)0, acc0, false, false);
        acc1 = __builtin_amdgcn_wmma_f32_16x16x32_f16(false, a, false, b1, (short)0, acc1, false, false);
        acc2 = __builtin_amdgcn_wmma_f32_16x16x32_f16(false, a, false, b2, (short)0, acc2, false, false);
        acc3 = __builtin_amdgcn_wmma_f32_16x16x32_f16(false, a, false, b3, (short)0, acc3, false, false);
    }

    // D layout: VGPR v -> M = v + 8*(lane>=16), N = lane%16 (+16*nt)
    float* row = dstage[wid];
    int mb = (lane & 16) ? 8 : 0;
    int nc = lane & 15;
    #pragma unroll
    for (int v = 0; v < 8; ++v) {
        row[(mb + v) * NCH +  0 + nc] = acc0[v];
        row[(mb + v) * NCH + 16 + nc] = acc1[v];
        row[(mb + v) * NCH + 32 + nc] = acc2[v];
        row[(mb + v) * NCH + 48 + nc] = acc3[v];
    }

    // coalesced b128 store: per-wave in-order DS guarantees RAW on `row`
    #pragma unroll
    for (int i = 0; i < 8; ++i) {
        int f  = i * 32 + lane;            // 256 float4 = 16 rows x 16 float4
        int m  = f >> 4;
        int c4 = f & 15;
        int jj = j0 + m;
        if (jj < HIN) {
            int ox = p + 2 * jj;
            v4f d = *(v4f*)&row[m * NCH + c4 * 4];
            *(v4f*)&votes[(((long)bic * H2 + oy) * H2 + ox) * NCH + c4 * 4] = d;
        }
    }
}

// ---------------------------------------------------------------------------
// Kernel 4: fused 3-iteration dynamic routing. One wave per output pixel.
// Lane holds elements (2l, 2l+1) of the 64 (oc,oa) values -> oc = lane/8.
// Votes cached in registers: read from HBM exactly once.
// ---------------------------------------------------------------------------
__global__ void __launch_bounds__(256) k_route(const float* __restrict__ votes,
                                               const float* __restrict__ bias,
                                               float* __restrict__ out) {
    const int lane = threadIdx.x & 31;
    const int wid  = threadIdx.x >> 5;
    int wv  = blockIdx.x * 8 + wid;        // 0..100351
    int b   = wv / NPIX;
    int pix = wv % NPIX;

    long base0 = (((long)b * ICN) * NPIX + pix) * NCH + 2 * lane;
    __builtin_prefetch((const void*)(votes + base0 + 8L * NCH), 0, 1); // global_prefetch_b8

    v2f vv[ICN];
    #pragma unroll
    for (int ic = 0; ic < ICN; ++ic)
        vv[ic] = *(const v2f*)(votes + base0 + (long)ic * NPIX * NCH);
    v2f bb = *(const v2f*)(bias + 2 * lane);

    float logit[ICN];
    #pragma unroll
    for (int ic = 0; ic < ICN; ++ic) logit[ic] = 0.f;

    float a0 = 0.f, a1 = 0.f;
    #pragma unroll
    for (int r = 0; r < 3; ++r) {
        float p0 = bb[0], p1 = bb[1];
        #pragma unroll
        for (int ic = 0; ic < ICN; ++ic) {
            // softmax over the 4 oc groups (8 lanes each): butterfly xor 8,16
            float li = logit[ic];
            float mx = fmaxf(li, __shfl_xor(li, 8, 32));
            float M  = fmaxf(mx, __shfl_xor(mx, 16, 32));
            float e  = __expf(li - M);
            float es = e + __shfl_xor(e, 8, 32);
            float tot = es + __shfl_xor(es, 16, 32);
            float route = e / tot;
            p0 = fmaf(route, vv[ic][0], p0);
            p1 = fmaf(route, vv[ic][1], p1);
        }
        // squash: ||v||^2 over the 16 oa values (8 lanes x 2): xor 1,2,4
        float ns = p0 * p0 + p1 * p1;
        ns += __shfl_xor(ns, 1, 32);
        ns += __shfl_xor(ns, 2, 32);
        ns += __shfl_xor(ns, 4, 32);
        float scale = __fsqrt_rn(ns) / (1.f + ns);   // v*norm/(1+norm^2)
        a0 = p0 * scale; a1 = p1 * scale;
        if (r < 2) {
            #pragma unroll
            for (int ic = 0; ic < ICN; ++ic) {
                float d = vv[ic][0] * a0 + vv[ic][1] * a1;
                d += __shfl_xor(d, 1, 32);
                d += __shfl_xor(d, 2, 32);
                d += __shfl_xor(d, 4, 32);
                logit[ic] += d;
            }
        }
    }
    v2f o; o[0] = a0; o[1] = a1;
    *(v2f*)(out + (((long)b * NPIX + pix) * NCH + 2 * lane)) = o;
}

// ---------------------------------------------------------------------------
extern "C" void kernel_launch(void* const* d_in, const int* in_sizes, int n_in,
                              void* d_out, int out_size, void* d_ws, size_t ws_size,
                              hipStream_t stream) {
    const float* x    = (const float*)d_in[0];   // (8,56,56,8,32)
    const float* W    = (const float*)d_in[1];   // (4,4,64,32)
    const float* bias = (const float*)d_in[2];   // (1,1,4,16)
    float* out = (float*)d_out;                  // (8,112,112,4,16)

    char* ws = (char*)d_ws;
    _Float16* xh    = (_Float16*)ws;                          // 6,422,528 f16 = 12,845,056 B
    _Float16* bfrag = (_Float16*)(ws + 12845056);             // 32,768 f16   = 65,536 B
    float*    votes = (float*)(ws + 12845056 + 65536);        // 51,380,224 f32

    k_pack_x<<<25088, 256, 0, stream>>>(x, xh);               // 6,422,528 / 256
    k_pack_w<<<128,   256, 0, stream>>>(W, bfrag);            // 32,768 / 256
    k_votes <<<7168,  256, 0, stream>>>(xh, bfrag, votes);    // 57,344 wave-tiles
    k_route <<<12544, 256, 0, stream>>>(votes, bias, out);    // 100,352 pixel-waves
}
